// NapCondCLAHELog_31258771980575
// MI455X (gfx1250) — matrix operand order repaired and checked by reading, not compiled, not run on hardware
//
#include <hip/hip_runtime.h>

typedef __attribute__((ext_vector_type(16))) _Float16 v16h;
typedef __attribute__((ext_vector_type(8)))  float    v8f;

#define NBINS 256
#define GH 8
#define GW 8
#define TH 128
#define TW 128
#define CCH 32
#define HH 1024
#define WW 1024
#define NPIX (CCH * HH * WW)           // 33554432
#define PIXELS (TH * TW)               // 16384 per tile
#define NTILES (CCH * GH * GW)         // 2048

// privatized histogram: 32 copies (8 waves x 4 lane-spread), padded stride 33
#define NCOPY   32
#define HSTRIDE 33
#define HTOTAL  (NBINS * HSTRIDE)      // 8448 entries = 33.8 KB

// ---- order-preserving float <-> uint key (for atomicMin/Max) ----
__device__ __forceinline__ unsigned fkey(float f) {
  unsigned b = __float_as_uint(f);
  return b ^ ((unsigned)(((int)b) >> 31) | 0x80000000u);
}
__device__ __forceinline__ float funkey(unsigned k) {
  unsigned b = (k & 0x80000000u) ? (k ^ 0x80000000u) : ~k;
  return __uint_as_float(b);
}

// ===================== kernel 0: init accumulators =====================
__global__ void k_init(double* acc, unsigned* keys) {
  acc[0] = 0.0;            // sum
  acc[1] = 0.0;            // sumsq
  keys[0] = 0xFFFFFFFFu;   // min key
  keys[1] = 0x00000000u;   // max key
}

// ===================== kernel 1: sum/sumsq/min/max =====================
__global__ void k_reduce(const float4* __restrict__ x4, double* acc, unsigned* keys) {
  __shared__ float    s_s[256], s_q[256];
  __shared__ unsigned s_mn[256], s_mx[256];
  const int tid = threadIdx.x;
  const long total  = NPIX / 4;
  const long stride = (long)gridDim.x * 256;
  long i = (long)blockIdx.x * 256 + tid;

  float s = 0.f, q = 0.f, mn = INFINITY, mx = -INFINITY;
  for (long p = i; p < total; p += stride) {
    __builtin_prefetch(&x4[p + stride], 0, 0);   // global_prefetch_b8
    float4 v = x4[p];
    float a0 = v.x, a1 = v.y, a2 = v.z, a3 = v.w;
    s += a0 + a1 + a2 + a3;
    q += a0 * a0 + a1 * a1 + a2 * a2 + a3 * a3;
    mn = fminf(mn, fminf(fminf(a0, a1), fminf(a2, a3)));
    mx = fmaxf(mx, fmaxf(fmaxf(a0, a1), fmaxf(a2, a3)));
  }
  s_s[tid] = s; s_q[tid] = q; s_mn[tid] = fkey(mn); s_mx[tid] = fkey(mx);
  for (int off = 128; off > 0; off >>= 1) {
    __syncthreads();
    if (tid < off) {
      s_s[tid] += s_s[tid + off];
      s_q[tid] += s_q[tid + off];
      s_mn[tid] = min(s_mn[tid], s_mn[tid + off]);
      s_mx[tid] = max(s_mx[tid], s_mx[tid + off]);
    }
  }
  if (tid == 0) {
    atomicAdd(&acc[0], (double)s_s[0]);   // global_atomic_add_f64
    atomicAdd(&acc[1], (double)s_q[0]);
    atomicMin(&keys[0], s_mn[0]);
    atomicMax(&keys[1], s_mx[0]);
  }
}

// ===================== kernel 2: scalars -> params =====================
// params: [0]=min, [1]=1/(max-min), [2]=max_val(clip), [3]=clahe flag
__global__ void k_scalars(const double* acc, const unsigned* keys, float* params) {
  const double n = (double)NPIX;
  double sum = acc[0], sq = acc[1];
  double var  = (sq - sum * sum / n) / (n - 1.0);   // unbiased, like torch .std()
  double sd   = sqrt(var);
  float mn = funkey(keys[0]);
  float mx = funkey(keys[1]);
  float range = mx - mn;
  float ct = (float)(sd / (double)range);           // ct_proxy = std/(max-min)
  int flag = ct < 0.05f;
  float max_val;
  if (flag) {
    float beta = 10.0f * ct;
    float cl   = 1.0f / (1.0f - beta) + beta;
    max_val = fmaxf(floorf(cl * (float)PIXELS / (float)NBINS), 1.0f);
  } else {
    max_val = (float)PIXELS;  // no clipping; LUT unused anyway
  }
  params[0] = mn;
  params[1] = 1.0f / range;
  params[2] = max_val;
  params[3] = flag ? 1.0f : 0.0f;
}

// ============ kernel 3: per-tile histogram + clip + WMMA scan -> LUT ============
// one 256-thread workgroup per tile; 32-way privatized LDS histogram (padded
// stride 33 -> bank-conflict-free update & merge); 256-bin inclusive cumsum as
// ONE v_wmma_f32_16x16x32_f16:  cum = [M | Lstrict](16x32) x [Uincl ; R](32x16)
__global__ void k_hist_lut(const float* __restrict__ x, const float* __restrict__ params,
                           float* __restrict__ g_lut) {
  __shared__ unsigned histc[HTOTAL];   // 32 copies, stride 33
  __shared__ float    fh[NBINS];       // clipped+redistributed histogram (16x16 matrix M)
  __shared__ float    red[NBINS];      // reduction scratch
  __shared__ float    rsum[16];        // row sums of M

  const int tid  = threadIdx.x;
  const int tile = blockIdx.x;               // (c, ty, tx) in C,GH,GW order
  const int c  = tile >> 6;
  const int ty = (tile >> 3) & 7;
  const int tx = tile & 7;

  const float mn  = params[0];
  const float inv = params[1];
  const float mv  = params[2];

  // copy index: wave id (tid>>5) x 4 lane-spread (tid&3) -> 0..31
  const int copy = (((tid >> 5) << 2) | (tid & 3));

  for (int i = tid; i < HTOTAL; i += 256) histc[i] = 0u;
  __syncthreads();

  // ---- histogram: 128x128 tile, float4 loads, privatized LDS atomics ----
  const float* base = x + (size_t)c * (HH * WW) + (size_t)(ty * TH) * WW + (size_t)tx * TW;
  const float4* b4 = (const float4*)base;
  #pragma unroll 4
  for (int it = 0; it < 16; ++it) {
    int qd   = it * 256 + tid;         // 0..4095 float4 slots
    int row  = qd >> 5;                // 128 rows
    int col4 = qd & 31;                // 32 float4 per row
    const float4* p = &b4[(size_t)row * (WW / 4) + col4];
    __builtin_prefetch(p + (WW / 4), 0, 0);
    float4 v = *p;
    float vs[4] = {v.x, v.y, v.z, v.w};
    int bb[4];
    #pragma unroll
    for (int j = 0; j < 4; ++j) {
      float x01 = fminf(fmaxf((vs[j] - mn) * inv, 0.0f), 1.0f);
      int b = (int)(x01 * (float)NBINS);           // trunc == floor (x01>=0)
      bb[j] = min(b, NBINS - 1);
    }
    // intra-thread dedup: smooth data -> all 4 pixels usually share one bin
    if (bb[0] == bb[1] && bb[0] == bb[2] && bb[0] == bb[3]) {
      atomicAdd(&histc[bb[0] * HSTRIDE + copy], 4u);
    } else {
      #pragma unroll
      for (int j = 0; j < 4; ++j) atomicAdd(&histc[bb[j] * HSTRIDE + copy], 1u);
    }
  }
  __syncthreads();

  // ---- merge 32 copies (stride 33 -> conflict-free reads) ----
  unsigned hcnt = 0u;
  #pragma unroll
  for (int cp = 0; cp < NCOPY; ++cp) hcnt += histc[tid * HSTRIDE + cp];

  // ---- clip at max_val, compute excess, redistribute ----
  float clipped = fminf((float)hcnt, mv);
  red[tid] = clipped;
  for (int off = 128; off > 0; off >>= 1) {
    __syncthreads();
    if (tid < off) red[tid] += red[tid + off];
  }
  __syncthreads();
  int ex = PIXELS - (int)red[0];       // exact: all values integral
  int residual = ex & (NBINS - 1);
  int redist   = ex >> 8;
  fh[tid] = clipped + (float)redist + ((tid < residual) ? 1.0f : 0.0f);
  __syncthreads();

  // ---- row sums of M (16x16 view of fh) ----
  if (tid < 16) {
    float r = 0.f;
    #pragma unroll
    for (int j = 0; j < 16; ++j) r += fh[tid * 16 + j];
    rsum[tid] = r;
  }
  __syncthreads();

  // ---- wave 0: single WMMA computes 256-element inclusive cumsum ----
  if (tid < 32) {                       // full wave -> EXEC all ones (WMMA requirement)
    const int lane  = tid;
    const int mrow  = lane & 15;
    const int kbase = (lane < 16) ? 0 : 8;    // A layout: lo lanes K=0..7,16..23; hi lanes +8
    v16h A = {};
    #pragma unroll
    for (int e = 0; e < 16; ++e) {
      int K = (e < 8) ? (kbase + e) : (16 + kbase + (e - 8));
      // A' = [ M (cols 0..15) | Lstrict (cols 16..31) ]
      float av = (K < 16) ? fh[mrow * 16 + K] : (((K - 16) < mrow) ? 1.0f : 0.0f);
      A[e] = (_Float16)av;
    }
    const int ncol = lane & 15;
    const int kb2  = (lane < 16) ? 0 : 8;     // B layout: lo lanes K=0..7,16..23; hi lanes +8
    v16h B = {};
    #pragma unroll
    for (int e = 0; e < 16; ++e) {
      int K = (e < 8) ? (kb2 + e) : (16 + kb2 + (e - 8));
      // B' = [ Uincl (rows 0..15) ; R rows: R[k][n] = rowsum_k ]
      float bv = (K < 16) ? ((K <= ncol) ? 1.0f : 0.0f) : rsum[K - 16];
      B[e] = (_Float16)bv;
    }
    v8f cz = {};
    v8f cum = __builtin_amdgcn_wmma_f32_16x16x32_f16(false, A, false, B, (short)0, cz,
                                                     false, false);
    // D layout: vgpr v, lanes 0-15 -> row v, lanes 16-31 -> row v+8; col = lane&15
    #pragma unroll
    for (int v = 0; v < 8; ++v) {
      int row = v + ((lane < 16) ? 0 : 8);
      int t = row * 16 + ncol;
      float l = floorf(fminf(fmaxf(cum[v] * ((float)(NBINS - 1) / (float)PIXELS), 0.0f),
                             (float)(NBINS - 1)));
      g_lut[(size_t)tile * NBINS + t] = l;
    }
  }
}

// ===================== kernel 4: bilinear LUT apply + log2 =====================
__global__ void k_apply(const float4* __restrict__ x4, const float* __restrict__ params,
                        const float* __restrict__ lut, float4* __restrict__ out4) {
  const size_t idx = (size_t)blockIdx.x * 256 + threadIdx.x;
  if (idx >= (size_t)NPIX / 4) return;

  const float mn  = params[0];
  const float inv = params[1];
  const bool flag = params[3] != 0.0f;

  float4 v = x4[idx];
  float xs[4] = {v.x, v.y, v.z, v.w};
  float os[4];

  const int c  = (int)(idx >> 18);          // 262144 float4 per channel
  const int q  = (int)(idx & 262143);
  const int h  = q >> 8;                    // 256 float4 per row
  const int w0 = (q & 255) * 4;

  if (flag) {
    float gy = ((float)h + 0.5f) * (1.0f / (float)TH) - 0.5f;
    float fy = floorf(gy);
    int y0 = min(max((int)fy, 0), GH - 1);
    int y1 = min(y0 + 1, GH - 1);
    float wy = fminf(fmaxf(gy - fy, 0.0f), 1.0f);
    const float* lutc = lut + (size_t)c * (GH * GW) * NBINS;
    #pragma unroll
    for (int j = 0; j < 4; ++j) {
      float x01 = fminf(fmaxf((xs[j] - mn) * inv, 0.0f), 1.0f);
      int b = min((int)(x01 * (float)(NBINS - 1)), NBINS - 1);
      float gx = ((float)(w0 + j) + 0.5f) * (1.0f / (float)TW) - 0.5f;
      float fx = floorf(gx);
      int xx0 = min(max((int)fx, 0), GW - 1);
      int xx1 = min(xx0 + 1, GW - 1);
      float wx = fminf(fmaxf(gx - fx, 0.0f), 1.0f);
      float v00 = lutc[(y0 * GW + xx0) * NBINS + b];
      float v01 = lutc[(y0 * GW + xx1) * NBINS + b];
      float v10 = lutc[(y1 * GW + xx0) * NBINS + b];
      float v11 = lutc[(y1 * GW + xx1) * NBINS + b];
      float top = v00 * (1.0f - wx) + v01 * wx;
      float bot = v10 * (1.0f - wx) + v11 * wx;
      float val = (top * (1.0f - wy) + bot * wy) * (1.0f / (float)(NBINS - 1));
      os[j] = __log2f(1.0f + val);
    }
  } else {
    #pragma unroll
    for (int j = 0; j < 4; ++j) {
      float x01 = fminf(fmaxf((xs[j] - mn) * inv, 0.0f), 1.0f);
      os[j] = __log2f(1.0f + x01);
    }
  }
  out4[idx] = make_float4(os[0], os[1], os[2], os[3]);
}

extern "C" void kernel_launch(void* const* d_in, const int* in_sizes, int n_in,
                              void* d_out, int out_size, void* d_ws, size_t ws_size,
                              hipStream_t stream) {
  (void)in_sizes; (void)n_in; (void)out_size; (void)ws_size;
  const float* x = (const float*)d_in[0];
  float* out = (float*)d_out;

  char* ws = (char*)d_ws;
  double*   acc    = (double*)ws;              // 16 B: sum, sumsq
  unsigned* keys   = (unsigned*)(ws + 16);     //  8 B: min/max keys
  float*    params = (float*)(ws + 32);        // 16 B: mn, 1/range, max_val, flag
  float*    lut    = (float*)(ws + 64);        // 2 MB: 2048 tiles x 256 bins

  k_init   <<<1,      1,   0, stream>>>(acc, keys);
  k_reduce <<<2048,   256, 0, stream>>>((const float4*)x, acc, keys);
  k_scalars<<<1,      1,   0, stream>>>(acc, keys, params);
  k_hist_lut<<<NTILES, 256, 0, stream>>>(x, params, lut);
  k_apply  <<<NPIX / 4 / 256, 256, 0, stream>>>((const float4*)x, params, lut,
                                                (float4*)out);
}
